// LundNet_23201413333516
// MI455X (gfx1250) — matrix operand order repaired and checked by reading, not compiled, not run on hardware
//
#include <hip/hip_runtime.h>

typedef __attribute__((ext_vector_type(16))) _Float16 v16h;
typedef __attribute__((ext_vector_type(8)))  _Float16 v8h;
typedef __attribute__((ext_vector_type(8)))  float    v8f;

#define NN   50000
#define NE   600000
#define NG   512
#define CATD 448
#define FUSED 384
#define FCD  256
#define EPS  1e-5f

union AF { v16h v; v8h h[2]; };

__device__ __forceinline__ long long ldIdx(const void* p, long long i, int is64) {
  return is64 ? ((const long long*)p)[i] : (long long)((const int*)p)[i];
}

__device__ __forceinline__ void atomAddF(float* p, float v) {
  unsafeAtomicAdd(p, v);
}

// ---------------------------------------------------------------- utilities
__global__ void detect_kernel(const void* ebuf, const void* bbuf, int* flags) {
  if (threadIdx.x != 0 || blockIdx.x != 0) return;
  const long long* e = (const long long*)ebuf;
  int ok = 1;
  for (int i = 0; i < 16; ++i) { long long v = e[i]; if (v < 0 || v >= NN) ok = 0; }
  flags[0] = ok;
  const long long* b = (const long long*)bbuf;
  int okb = 1;
  for (int i = 0; i < 8; ++i) { long long v = b[NN/2 - 1 - i]; if (v < 0 || v >= NG) okb = 0; }
  flags[1] = okb;
}

// s = g*rsqrt(v+eps); t = b - m*s (+ bias*s if foldbias)
__global__ void fold_bn_kernel(const float* __restrict__ g, const float* __restrict__ b,
                               const float* __restrict__ m, const float* __restrict__ v,
                               const float* __restrict__ bias, int foldbias, int n,
                               float* __restrict__ S, float* __restrict__ T) {
  int i = blockIdx.x * blockDim.x + threadIdx.x;
  if (i < n) {
    float s = g[i] * rsqrtf(v[i] + EPS);
    float t = b[i] - m[i] * s;
    if (foldbias) t += bias[i] * s;
    S[i] = s; T[i] = t;
  }
}

__global__ void repack_kernel(const float* __restrict__ src, _Float16* __restrict__ dst, int n) {
  int i = blockIdx.x * blockDim.x + threadIdx.x;
  if (i < n) dst[i] = (_Float16)src[i];
}

__global__ void bn_input_kernel(const float* __restrict__ x,
                                const float* __restrict__ g, const float* __restrict__ b,
                                const float* __restrict__ m, const float* __restrict__ v,
                                _Float16* __restrict__ h0) {
  int i = blockIdx.x * blockDim.x + threadIdx.x;
  if (i < NN * 3) {
    int node = i / 3, c = i - node * 3;
    float s = g[c] * rsqrtf(v[c] + EPS);
    h0[(size_t)node * 4 + c] = (_Float16)((x[i] - m[c]) * s + b[c]);
  }
}

__global__ void deg_kernel(const void* __restrict__ ebuf, const int* __restrict__ flags,
                           float* __restrict__ deg) {
  const int is64 = flags[0];
  int i = blockIdx.x * blockDim.x + threadIdx.x;
  if (i < NE) atomAddF(&deg[(int)ldIdx(ebuf, (long long)NE + i, is64)], 1.0f);
}

__global__ void gcnt_kernel(const void* __restrict__ bbuf, const int* __restrict__ flags,
                            float* __restrict__ gcnt) {
  const int is64 = flags[1];
  int i = blockIdx.x * blockDim.x + threadIdx.x;
  if (i < NN) atomAddF(&gcnt[(int)ldIdx(bbuf, i, is64)], 1.0f);
}

__global__ void finalize_kernel(const float* __restrict__ num, const float* __restrict__ deg,
                                _Float16* __restrict__ hcat, int colOff, int cout) {
  int i = blockIdx.x * blockDim.x + threadIdx.x;
  if (i < NN * cout) {
    int node = i / cout, c = i - node * cout;
    float d = deg[node]; d = d < 1.f ? 1.f : d;
    hcat[(size_t)node * CATD + colOff + c] = (_Float16)(num[i] / d);
  }
}

__global__ void pooldiv_kernel(const float* __restrict__ gsum, const float* __restrict__ gcnt,
                               _Float16* __restrict__ g16) {
  int i = blockIdx.x * blockDim.x + threadIdx.x;
  if (i < NG * FUSED) {
    int g = i / FUSED;
    float c = gcnt[g]; c = c < 1.f ? 1.f : c;
    g16[i] = (_Float16)(gsum[i] / c);
  }
}

// ---------------------------------------------------------------- fused EdgeConv layer
// per 16-edge tile: gather [xi | xj-xi] -> WMMA mm1 -> bn1+relu -> LDS -> WMMA mm2
//                   -> +b2, relu, bn2 -> global f32 atomic scatter into num[dst]
template<int K1, int C_IN, int C_OUT>
__launch_bounds__(128, 2)
__global__ void conv_layer_kernel(const _Float16* __restrict__ hin, int in_stride,
                                  const void* __restrict__ ebuf, const int* __restrict__ flags,
                                  const float* __restrict__ w1, const float* __restrict__ w2,
                                  const float* __restrict__ A1, const float* __restrict__ B1,
                                  const float* __restrict__ b2,
                                  const float* __restrict__ A2, const float* __restrict__ B2,
                                  float* __restrict__ num) {
  constexpr int NT  = C_OUT / 16;
  constexpr int W1E = K1 * C_OUT;
  constexpr int W2E = C_OUT * C_OUT;
  extern __shared__ _Float16 smem[];
  _Float16* w1s = smem;                      // K1 x C_OUT, row-major, zero-padded rows
  _Float16* w2s = w1s + W1E;                 // C_OUT x C_OUT
  _Float16* msg = w2s + W2E;                 // 4 waves x 16 x K1
  _Float16* m1s = msg + 4 * 16 * K1;         // 4 waves x 16 x C_OUT
  int*      dsts = (int*)(m1s + 4 * 16 * C_OUT); // 4 waves x 16

  const int tid = threadIdx.x;
  const int is64 = flags[0];

  for (int i = tid; i < W1E; i += 128) {
    int r = i / C_OUT;
    w1s[i] = (r < 2 * C_IN) ? (_Float16)w1[i] : (_Float16)0.f;
  }
  for (int i = tid; i < W2E; i += 128) w2s[i] = (_Float16)w2[i];
  __syncthreads();

  const int wave = tid >> 5;
  const int lane = tid & 31;
  const int lrow = lane & 15;
  const int hi   = lane >> 4;
  const int rb   = hi * 8;
  _Float16* mymsg = msg + wave * 16 * K1;
  _Float16* mym1  = m1s + wave * 16 * C_OUT;
  int* mydst = dsts + wave * 16;

  const int NTILES = NE / 16;                 // 37500, divides (gridDim*4) exactly
  const int wtot = gridDim.x * 4;
  v8f zf = {};

  for (int tile = blockIdx.x * 4 + wave; tile < NTILES; tile += wtot) {
    const int e0 = tile * 16;
    // ---- gather ----
    {
      const int e = e0 + lrow;
      const long long di = ldIdx(ebuf, (long long)NE + e, is64);
      const _Float16* xi = hin + (size_t)di * in_stride;
      if (hi == 0) {
        mydst[lrow] = (int)di;
        if constexpr (C_IN % 8 == 0) {
          for (int c = 0; c < C_IN; c += 8)
            *(v8h*)(mymsg + lrow * K1 + c) = *(const v8h*)(xi + c);
        } else {
          for (int c = 0; c < C_IN; ++c) mymsg[lrow * K1 + c] = xi[c];
        }
      } else {
        const long long sj = ldIdx(ebuf, e, is64);
        const _Float16* xj = hin + (size_t)sj * in_stride;
        if constexpr (C_IN % 8 == 0) {
          for (int c = 0; c < C_IN; c += 8) {
            v8h a = *(const v8h*)(xj + c);
            v8h b = *(const v8h*)(xi + c);
            *(v8h*)(mymsg + lrow * K1 + C_IN + c) = a - b;
          }
        } else {
          for (int c = 0; c < C_IN; ++c)
            mymsg[lrow * K1 + C_IN + c] = (_Float16)((float)xj[c] - (float)xi[c]);
          for (int c = 2 * C_IN; c < K1; ++c) mymsg[lrow * K1 + c] = (_Float16)0.f;
        }
      }
    }
    __syncthreads();

    // ---- mm1: relu(bn1(msg @ w1 + b1))   (bias folded into B1) ----
    v8f acc[NT];
    #pragma unroll
    for (int nt = 0; nt < NT; ++nt) acc[nt] = zf;
    #pragma clang loop unroll_count(2)
    for (int kc = 0; kc < K1; kc += 32) {
      AF a;
      const _Float16* ab = mymsg + lrow * K1 + kc + hi * 8;
      a.h[0] = *(const v8h*)ab;
      a.h[1] = *(const v8h*)(ab + 16);
      #pragma unroll
      for (int nt = 0; nt < NT; ++nt) {
        v16h b;
        for (int s = 0; s < 16; ++s) b[s] = w1s[(kc + hi * 8 + (s & 7) + (s >> 3) * 16) * C_OUT + nt * 16 + lrow];
        acc[nt] = __builtin_amdgcn_wmma_f32_16x16x32_f16(false, a.v, false, b,
                                                         (short)0, acc[nt], false, false);
      }
    }
    #pragma unroll
    for (int nt = 0; nt < NT; ++nt) {
      const int n = nt * 16 + lrow;
      const float s = A1[n], t = B1[n];
      #pragma unroll
      for (int r = 0; r < 8; ++r) {
        float y = acc[nt][r] * s + t;
        y = y > 0.f ? y : 0.f;
        mym1[(rb + r) * C_OUT + n] = (_Float16)y;
      }
    }
    __syncthreads();

    // ---- mm2: bn2(relu(m1 @ w2 + b2)) -> atomic scatter ----
    #pragma unroll
    for (int nt = 0; nt < NT; ++nt) acc[nt] = zf;
    #pragma clang loop unroll_count(2)
    for (int kc = 0; kc < C_OUT; kc += 32) {
      AF a;
      const _Float16* ab = mym1 + lrow * C_OUT + kc + hi * 8;
      a.h[0] = *(const v8h*)ab;
      a.h[1] = *(const v8h*)(ab + 16);
      #pragma unroll
      for (int nt = 0; nt < NT; ++nt) {
        v16h b;
        for (int s = 0; s < 16; ++s) b[s] = w2s[(kc + hi * 8 + (s & 7) + (s >> 3) * 16) * C_OUT + nt * 16 + lrow];
        acc[nt] = __builtin_amdgcn_wmma_f32_16x16x32_f16(false, a.v, false, b,
                                                         (short)0, acc[nt], false, false);
      }
    }
    #pragma unroll
    for (int nt = 0; nt < NT; ++nt) {
      const int n = nt * 16 + lrow;
      const float bb = b2[n], s = A2[n], t = B2[n];
      #pragma unroll
      for (int r = 0; r < 8; ++r) {
        float z = acc[nt][r] + bb;
        z = z > 0.f ? z : 0.f;
        atomAddF(num + (size_t)mydst[rb + r] * C_OUT + n, z * s + t);
      }
    }
    __syncthreads();
  }
}

// ---------------------------------------------------------------- fusion (448->384) + mean-pool scatter
__launch_bounds__(128, 4)
__global__ void fusion_pool_kernel(const _Float16* __restrict__ hcat,
                                   const _Float16* __restrict__ w16,   // [448][384] f16
                                   const float* __restrict__ bias,
                                   const float* __restrict__ S, const float* __restrict__ T,
                                   const void* __restrict__ bbuf, const int* __restrict__ flags,
                                   float* __restrict__ gsum) {
  const int tid = threadIdx.x;
  const int wave = tid >> 5, lane = tid & 31;
  const int lrow = lane & 15, hi = lane >> 4, rb = hi * 8;
  const int is64 = flags[1];
  const int ITEMS = (NN / 16) * 4;
  const int wtot = gridDim.x * 4;
  v8f zf = {};
  for (int it = blockIdx.x * 4 + wave; it < ITEMS; it += wtot) {
    const int ntile = it >> 2, cg = it & 3;
    const int node0 = ntile * 16;
    const int c0 = cg * 96;
    int bid[8];
    #pragma unroll
    for (int r = 0; r < 8; ++r) bid[r] = (int)ldIdx(bbuf, node0 + rb + r, is64);
    v8f acc[6];
    #pragma unroll
    for (int j = 0; j < 6; ++j) acc[j] = zf;
    #pragma clang loop unroll_count(2)
    for (int kc = 0; kc < CATD; kc += 32) {
      AF a;
      const _Float16* ab = hcat + (size_t)(node0 + lrow) * CATD + kc + hi * 8;
      a.h[0] = *(const v8h*)ab;
      a.h[1] = *(const v8h*)(ab + 16);
      #pragma unroll
      for (int j = 0; j < 6; ++j) {
        v16h b;
        for (int s = 0; s < 16; ++s) b[s] = w16[(size_t)(kc + hi * 8 + (s & 7) + (s >> 3) * 16) * FUSED + c0 + j * 16 + lrow];
        acc[j] = __builtin_amdgcn_wmma_f32_16x16x32_f16(false, a.v, false, b,
                                                        (short)0, acc[j], false, false);
      }
    }
    #pragma unroll
    for (int j = 0; j < 6; ++j) {
      const int n = c0 + j * 16 + lrow;
      const float bb = bias[n], s = S[n], t = T[n];
      #pragma unroll
      for (int r = 0; r < 8; ++r) {
        float z = acc[j][r] + bb; z = z > 0.f ? z : 0.f;
        atomAddF(gsum + (size_t)bid[r] * FUSED + n, z * s + t);
      }
    }
  }
}

// ---------------------------------------------------------------- FC 384->256 (relu+bn)
__launch_bounds__(128, 4)
__global__ void fc1_kernel(const _Float16* __restrict__ g16,   // [512][384]
                           const _Float16* __restrict__ w16,   // [384][256]
                           const float* __restrict__ bias,
                           const float* __restrict__ S, const float* __restrict__ T,
                           _Float16* __restrict__ out16) {     // [512][256]
  const int tid = threadIdx.x;
  const int wave = tid >> 5, lane = tid & 31;
  const int lrow = lane & 15, hi = lane >> 4, rb = hi * 8;
  const int it = blockIdx.x * 4 + wave;      // 32 gtiles x 4 col groups = 128 waves
  const int gtile = it >> 2, cg = it & 3;
  const int g0 = gtile * 16, c0 = cg * 64;
  v8f zf = {};
  v8f acc[4];
  #pragma unroll
  for (int j = 0; j < 4; ++j) acc[j] = zf;
  #pragma clang loop unroll_count(2)
  for (int kc = 0; kc < FUSED; kc += 32) {
    AF a;
    const _Float16* ab = g16 + (size_t)(g0 + lrow) * FUSED + kc + hi * 8;
    a.h[0] = *(const v8h*)ab;
    a.h[1] = *(const v8h*)(ab + 16);
    #pragma unroll
    for (int j = 0; j < 4; ++j) {
      v16h b;
      for (int s = 0; s < 16; ++s) b[s] = w16[(size_t)(kc + hi * 8 + (s & 7) + (s >> 3) * 16) * FCD + c0 + j * 16 + lrow];
      acc[j] = __builtin_amdgcn_wmma_f32_16x16x32_f16(false, a.v, false, b,
                                                      (short)0, acc[j], false, false);
    }
  }
  #pragma unroll
  for (int j = 0; j < 4; ++j) {
    const int n = c0 + j * 16 + lrow;
    const float bb = bias[n], s = S[n], t = T[n];
    #pragma unroll
    for (int r = 0; r < 8; ++r) {
      float z = acc[j][r] + bb; z = z > 0.f ? z : 0.f;
      out16[(size_t)(g0 + rb + r) * FCD + n] = (_Float16)(z * s + t);
    }
  }
}

// ---------------------------------------------------------------- out layer 256->1, bn, sigmoid
__global__ void out_kernel(const _Float16* __restrict__ g16,  // [512][256]
                           const float* __restrict__ w,       // [256]
                           const float* __restrict__ S, const float* __restrict__ T,
                           float* __restrict__ out) {
  const int wave = threadIdx.x >> 5, lane = threadIdx.x & 31;
  const int g = blockIdx.x * 4 + wave;
  if (g >= NG) return;
  float v = 0.f;
  for (int c = lane; c < FCD; c += 32)
    v += (float)g16[(size_t)g * FCD + c] * w[c];
  for (int off = 16; off > 0; off >>= 1) v += __shfl_down(v, off, 32);
  if (lane == 0) {
    float y = v * S[0] + T[0];     // bias folded into T
    out[g] = 1.f / (1.f + __expf(-y));
  }
}

// ---------------------------------------------------------------- host
extern "C" void kernel_launch(void* const* d_in, const int* in_sizes, int n_in,
                              void* d_out, int out_size, void* d_ws, size_t ws_size,
                              hipStream_t stream) {
  (void)n_in; (void)out_size; (void)ws_size;
  const bool sortedOrder = (in_sizes[0] == NN);  // tree-flatten w/ sorted dict keys
  auto F = [&](int i) { return (const float*)d_in[i]; };

  const float* xin = nullptr; const void* ebuf = nullptr; const void* bbuf = nullptr;
  const float *bnf_g, *bnf_b, *bnf_m, *bnf_v;
  struct ConvP { const float *w1,*b1,*g1,*be1,*m1,*v1,*w2,*b2,*g2,*be2,*m2,*v2; } cv[6];
  const float *fw,*fb,*fg,*fbe,*fm,*fv;         // fusion
  const float *cw,*cb,*cg_,*cbe,*cm,*cvv;       // fc
  const float *ow,*ob,*og,*obe,*om,*ov;         // out

  if (!sortedOrder) {  // dict insertion order: x, edge_index, batch, params{...}
    xin = F(0); ebuf = d_in[1]; bbuf = d_in[2];
    bnf_g = F(3); bnf_b = F(4); bnf_m = F(5); bnf_v = F(6);
    int p = 7;
    for (int l = 0; l < 6; ++l, p += 12)
      cv[l] = { F(p),F(p+1),F(p+2),F(p+3),F(p+4),F(p+5),F(p+6),F(p+7),F(p+8),F(p+9),F(p+10),F(p+11) };
    fw=F(p); fb=F(p+1); fg=F(p+2); fbe=F(p+3); fm=F(p+4); fv=F(p+5); p += 6;
    cw=F(p); cb=F(p+1); cg_=F(p+2); cbe=F(p+3); cm=F(p+4); cvv=F(p+5); p += 6;
    ow=F(p); ob=F(p+1); og=F(p+2); obe=F(p+3); om=F(p+4); ov=F(p+5);
  } else {             // sorted keys: batch, edge_index, params{bn_fts,convs,fcs,fusion,out}, x
    bbuf = d_in[0]; ebuf = d_in[1];
    bnf_b = F(2); bnf_g = F(3); bnf_m = F(4); bnf_v = F(5);
    int p = 6;
    for (int l = 0; l < 6; ++l, p += 12) {
      cv[l].b1=F(p);   cv[l].b2=F(p+1);
      cv[l].be1=F(p+2); cv[l].g1=F(p+3); cv[l].m1=F(p+4); cv[l].v1=F(p+5);
      cv[l].be2=F(p+6); cv[l].g2=F(p+7); cv[l].m2=F(p+8); cv[l].v2=F(p+9);
      cv[l].w1=F(p+10); cv[l].w2=F(p+11);
    }
    cb=F(p); cbe=F(p+1); cg_=F(p+2); cm=F(p+3); cvv=F(p+4); cw=F(p+5); p += 6;
    fb=F(p); fbe=F(p+1); fg=F(p+2); fm=F(p+3); fv=F(p+4); fw=F(p+5); p += 6;
    ob=F(p); obe=F(p+1); og=F(p+2); om=F(p+3); ov=F(p+4); ow=F(p+5); p += 6;
    xin = F(p);
  }

  // workspace carve
  char* wp = (char*)d_ws;
  auto carve = [&](size_t bytes) -> char* {
    char* r = wp; wp += (bytes + 255) & ~(size_t)255; return r;
  };
  _Float16* hcat = (_Float16*)carve((size_t)NN * CATD * 2);
  _Float16* h0   = (_Float16*)carve((size_t)NN * 4 * 2);
  float* num  = (float*)carve((size_t)NN * 128 * 4);
  float* deg  = (float*)carve((size_t)NN * 4);
  float* gsum = (float*)carve((size_t)NG * FUSED * 4);
  float* gcnt = (float*)carve((size_t)NG * 4);
  _Float16* gf   = (_Float16*)carve((size_t)NG * FUSED * 2);
  _Float16* g2f  = (_Float16*)carve((size_t)NG * FCD * 2);
  _Float16* fw16 = (_Float16*)carve((size_t)CATD * FUSED * 2);
  _Float16* cw16 = (_Float16*)carve((size_t)FUSED * FCD * 2);
  float* fold = (float*)carve(8192 * 4);
  int* flags  = (int*)carve(256);

  // fold BN params
  const int CO[6] = {32, 32, 64, 64, 128, 128};
  for (int l = 0; l < 6; ++l) {
    float *A1 = fold + l * 512, *B1 = A1 + 128, *A2 = A1 + 256, *B2 = A1 + 384;
    fold_bn_kernel<<<1, 128, 0, stream>>>(cv[l].g1, cv[l].be1, cv[l].m1, cv[l].v1, cv[l].b1, 1, CO[l], A1, B1);
    fold_bn_kernel<<<1, 128, 0, stream>>>(cv[l].g2, cv[l].be2, cv[l].m2, cv[l].v2, nullptr, 0, CO[l], A2, B2);
  }
  float *fS = fold + 3072, *fT = fold + 3456, *cS = fold + 3840, *cT = fold + 4096,
        *oS = fold + 4352, *oT = fold + 4353;
  fold_bn_kernel<<<3, 128, 0, stream>>>(fg, fbe, fm, fv, nullptr, 0, FUSED, fS, fT);
  fold_bn_kernel<<<2, 128, 0, stream>>>(cg_, cbe, cm, cvv, nullptr, 0, FCD, cS, cT);
  fold_bn_kernel<<<1, 32, 0, stream>>>(og, obe, om, ov, ob, 1, 1, oS, oT);

  detect_kernel<<<1, 32, 0, stream>>>(ebuf, bbuf, flags);
  repack_kernel<<<(CATD * FUSED + 255) / 256, 256, 0, stream>>>(fw, fw16, CATD * FUSED);
  repack_kernel<<<(FUSED * FCD + 255) / 256, 256, 0, stream>>>(cw, cw16, FUSED * FCD);
  bn_input_kernel<<<(NN * 3 + 255) / 256, 256, 0, stream>>>(xin, bnf_g, bnf_b, bnf_m, bnf_v, h0);
  hipMemsetAsync(deg, 0, (size_t)NN * 4, stream);
  deg_kernel<<<(NE + 255) / 256, 256, 0, stream>>>(ebuf, flags, deg);

  auto smemSz = [](int K1, int COUT) -> size_t {
    return (size_t)(K1 * COUT + COUT * COUT + 64 * K1 + 64 * COUT) * 2 + 64 * sizeof(int);
  };
  const int colOff[6] = {0, 32, 64, 128, 192, 320};
  for (int l = 0; l < 6; ++l) {
    float *A1 = fold + l * 512, *B1 = A1 + 128, *A2 = A1 + 256, *B2 = A1 + 384;
    hipMemsetAsync(num, 0, (size_t)NN * CO[l] * 4, stream);
    const _Float16* hin = (l == 0) ? h0 : (hcat + colOff[l - 1]);
    int stride = (l == 0) ? 4 : CATD;
    switch (l) {
      case 0: conv_layer_kernel<32, 3, 32><<<625, 128, smemSz(32, 32), stream>>>(
                  hin, stride, ebuf, flags, cv[0].w1, cv[0].w2, A1, B1, cv[0].b2, A2, B2, num); break;
      case 1: conv_layer_kernel<64, 32, 32><<<625, 128, smemSz(64, 32), stream>>>(
                  hin, stride, ebuf, flags, cv[1].w1, cv[1].w2, A1, B1, cv[1].b2, A2, B2, num); break;
      case 2: conv_layer_kernel<64, 32, 64><<<625, 128, smemSz(64, 64), stream>>>(
                  hin, stride, ebuf, flags, cv[2].w1, cv[2].w2, A1, B1, cv[2].b2, A2, B2, num); break;
      case 3: conv_layer_kernel<128, 64, 64><<<625, 128, smemSz(128, 64), stream>>>(
                  hin, stride, ebuf, flags, cv[3].w1, cv[3].w2, A1, B1, cv[3].b2, A2, B2, num); break;
      case 4: conv_layer_kernel<128, 64, 128><<<625, 128, smemSz(128, 128), stream>>>(
                  hin, stride, ebuf, flags, cv[4].w1, cv[4].w2, A1, B1, cv[4].b2, A2, B2, num); break;
      case 5: conv_layer_kernel<256, 128, 128><<<625, 128, smemSz(256, 128), stream>>>(
                  hin, stride, ebuf, flags, cv[5].w1, cv[5].w2, A1, B1, cv[5].b2, A2, B2, num); break;
    }
    finalize_kernel<<<(NN * CO[l] + 255) / 256, 256, 0, stream>>>(num, deg, hcat, colOff[l], CO[l]);
  }

  hipMemsetAsync(gsum, 0, (size_t)NG * FUSED * 4, stream);
  hipMemsetAsync(gcnt, 0, (size_t)NG * 4, stream);
  gcnt_kernel<<<(NN + 255) / 256, 256, 0, stream>>>(bbuf, flags, gcnt);
  fusion_pool_kernel<<<1024, 128, 0, stream>>>(hcat, fw16, fb, fS, fT, bbuf, flags, gsum);
  pooldiv_kernel<<<(NG * FUSED + 255) / 256, 256, 0, stream>>>(gsum, gcnt, gf);
  fc1_kernel<<<32, 128, 0, stream>>>(gf, cw16, cb, cS, cT, g2f);
  out_kernel<<<NG / 4, 128, 0, stream>>>(g2f, ow, oS, oT, (float*)d_out);
}
